// GRU_53824530153963
// MI455X (gfx1250) — matrix operand order repaired and checked by reading, not compiled, not run on hardware
//
#include <hip/hip_runtime.h>

// ---------------------------------------------------------------------------
// CDNA5 / gfx1250 GRU-LM:  bf16 WMMA GEMMs (v_wmma_f32_16x16x32_bf16) with
// async direct-to-LDS tile staging (global_load_async_to_lds_b128, ASYNCcnt)
// and double-buffered LDS. Layer-0 x-projections hoisted out of the scan,
// decoder hoisted into one big [4096 x 32000 x 1024] GEMM.
// ---------------------------------------------------------------------------

typedef __bf16 bf16_t;
typedef __attribute__((ext_vector_type(16))) __bf16 v16bf;
typedef __attribute__((ext_vector_type(8)))  __bf16 v8bf;
typedef __attribute__((ext_vector_type(8)))  float  v8f;

#define ACT_NONE    0
#define ACT_SIGMOID 1
#define ACT_TANH    2

// CDNA5 async copy: DMA 16B per lane straight into LDS, tracked by ASYNCcnt.
// (no stable clang builtin across toolchains -> inline asm per bridge doc)
__device__ __forceinline__ void async_load_b128(unsigned lds_addr, const void* gaddr) {
    asm volatile("global_load_async_to_lds_b128 %0, %1, off"
                 :: "v"(lds_addr), "v"(gaddr) : "memory");
}
__device__ __forceinline__ void wait_async0() {
    asm volatile("s_wait_asynccnt 0x0" ::: "memory");
}
// flat pointer -> LDS byte offset (ISA 10.2: LDS aperture addr = addr[31:0])
__device__ __forceinline__ unsigned lds_off(const void* p) {
    return (unsigned)(uintptr_t)p;
}

// Generic GEMM: Cout[M,N] = act( A[M,K](bf16) * W[N,K](bf16)^T + Cinit + bias )
// Tiling: 64x64 per workgroup, 128 threads = 4 waves, each wave owns a
// 16(M) x 64(N) strip -> 4 wmma accumulators. K chunked by 32, double-buffered.
__global__ __launch_bounds__(128)
void gemm_bf16_wmma(const bf16_t* __restrict__ A,
                    const bf16_t* __restrict__ W,
                    const float*  __restrict__ Cinit,
                    const float*  __restrict__ bias,
                    float*        __restrict__ Cout,
                    int M, int N, int K, int act)
{
    // padded rows (40*2B = 80B stride) to avoid LDS bank conflicts
    __shared__ bf16_t As[2][64][40];
    __shared__ bf16_t Bs[2][64][40];

    const int n0   = blockIdx.x * 64;
    const int m0   = blockIdx.y * 64;
    const int tid  = threadIdx.x;
    const int wave = tid >> 5;        // 0..3  -> M sub-tile
    const int lane = tid & 31;
    const int l15  = lane & 15;
    const int hi   = lane >> 4;       // 0 / 1 : lane half
    const int lrow = tid >> 1;        // 0..63 staging row
    const int lcol = (tid & 1) * 16;  // staging K-half

    v8f acc[4] = {};

    const bf16_t* gArow = A + (size_t)(m0 + lrow) * K + lcol;
    const bf16_t* gBrow = W + (size_t)(n0 + lrow) * K + lcol;

    // issue async DMA of one 64x32 A tile + 64x32 W tile into LDS buffer `buf`
    auto stage = [&](int buf, int k0) {
        const bf16_t* ga = gArow + k0;
        const bf16_t* gb = gBrow + k0;
        unsigned sa = lds_off(&As[buf][lrow][lcol]);
        unsigned sb = lds_off(&Bs[buf][lrow][lcol]);
        async_load_b128(sa,      ga);
        async_load_b128(sa + 16, (const char*)ga + 16);
        async_load_b128(sb,      gb);
        async_load_b128(sb + 16, (const char*)gb + 16);
    };

    stage(0, 0);
    int cur = 0;

    for (int k0 = 0; k0 < K; k0 += 32) {
        wait_async0();        // my wave's DMA into `cur` has landed in LDS
        __syncthreads();      // ...and everyone else's too; prev reads done
        if (k0 + 32 < K) stage(cur ^ 1, k0 + 32);   // overlap next copy

        // ---- A fragment (16x32, ISA 7.12.2): lane l<16 -> row l, K 0-7 & 16-23
        //                                      lane l+16 -> row l, K 8-15 & 24-31
        const int arow = 16 * wave + l15;
        v8bf a0 = *(const v8bf*)(&As[cur][arow][8 * hi]);
        v8bf a1 = *(const v8bf*)(&As[cur][arow][16 + 8 * hi]);
        v16bf afrag = __builtin_shufflevector(a0, a1,
            0,1,2,3,4,5,6,7,8,9,10,11,12,13,14,15);

        // ---- 4 N-tiles: B fragment lane n<16 -> col n, K 0-7 & 16-23 ; lane n+16 -> K 8-15 & 24-31
#pragma unroll
        for (int nt = 0; nt < 4; ++nt) {
            const int brow = 16 * nt + l15;
            v8bf b0 = *(const v8bf*)(&Bs[cur][brow][8 * hi]);
            v8bf b1 = *(const v8bf*)(&Bs[cur][brow][16 + 8 * hi]);
            v16bf bfrag = __builtin_shufflevector(b0, b1,
                0,1,2,3,4,5,6,7,8,9,10,11,12,13,14,15);
            acc[nt] = __builtin_amdgcn_wmma_f32_16x16x32_bf16(
                false, afrag, false, bfrag, (short)0, acc[nt], false, false);
        }
        cur ^= 1;
    }

    // ---- epilogue: D layout vgpr r -> row r + 8*hi, col = lane&15
#pragma unroll
    for (int nt = 0; nt < 4; ++nt) {
        const int col = n0 + 16 * nt + l15;
        const float bv = bias ? bias[col] : 0.0f;
#pragma unroll
        for (int r = 0; r < 8; ++r) {
            const int row = m0 + 16 * wave + r + 8 * hi;
            float v = acc[nt][r] + bv;
            if (Cinit) v += Cinit[(size_t)row * N + col];
            if (act == ACT_SIGMOID)   v = 1.0f / (1.0f + __expf(-v));
            else if (act == ACT_TANH) v = tanhf(v);
            Cout[(size_t)row * N + col] = v;
        }
    }
}

// ---------------------------------------------------------------------------
// elementwise helpers
// ---------------------------------------------------------------------------
__global__ void cvt_f32_bf16(const float* __restrict__ s, bf16_t* __restrict__ d, int n) {
    int i = blockIdx.x * 256 + threadIdx.x;
    if (i < n) d[i] = (bf16_t)s[i];
}

__global__ void embed_gather(const int* __restrict__ tok, const float* __restrict__ emb,
                             bf16_t* __restrict__ x, int n, int E) {
    int i = blockIdx.x * 256 + threadIdx.x;
    if (i < n) {
        int r = i / E, c = i - r * E;
        x[i] = (bf16_t)emb[(size_t)tok[r] * E + c];
    }
}

__global__ void init_h(const float* __restrict__ h0, float* __restrict__ hf,
                       bf16_t* __restrict__ hb, int n) {
    int i = blockIdx.x * 256 + threadIdx.x;
    if (i < n) { float v = h0[i]; hf[i] = v; hb[i] = (bf16_t)v; }
}

__global__ void rh_mul(const float* __restrict__ r, const float* __restrict__ h,
                       bf16_t* __restrict__ rh, int n) {
    int i = blockIdx.x * 256 + threadIdx.x;
    if (i < n) rh[i] = (bf16_t)(r[i] * h[i]);
}

__global__ void gru_update(const float* __restrict__ z, const float* __restrict__ hc,
                           float* __restrict__ hf, bf16_t* __restrict__ hb,
                           bf16_t* __restrict__ hist, int n) {
    int i = blockIdx.x * 256 + threadIdx.x;
    if (i < n) {
        float zz = z[i];
        float hn = (1.0f - zz) * hf[i] + zz * hc[i];
        hf[i] = hn;
        bf16_t b = (bf16_t)hn;
        hb[i] = b;
        if (hist) hist[i] = b;
    }
}

__global__ void copy_f32(const float* __restrict__ s, float* __restrict__ d, int n) {
    int i = blockIdx.x * 256 + threadIdx.x;
    if (i < n) d[i] = s[i];
}

// ---------------------------------------------------------------------------
extern "C" void kernel_launch(void* const* d_in, const int* in_sizes, int n_in,
                              void* d_out, int out_size, void* d_ws, size_t ws_size,
                              hipStream_t stream)
{
    (void)in_sizes; (void)n_in; (void)out_size; (void)ws_size;
    constexpr int L = 2, B = 64, S = 64, E = 1024, H = 1024, V = 32000;
    constexpr size_t HH  = (size_t)H * H;
    constexpr int    BH  = B * H;          // 65536
    constexpr int    SBH = S * B * H;      // 4,194,304

    const int*   inputs = (const int*)  d_in[0];
    const float* hidden = (const float*)d_in[1];
    const float* emb    = (const float*)d_in[2];
    const float* Wx = (const float*)d_in[3];
    const float* Ur = (const float*)d_in[4];
    const float* br = (const float*)d_in[5];
    const float* Wz = (const float*)d_in[6];
    const float* Uz = (const float*)d_in[7];
    const float* bz = (const float*)d_in[8];
    const float* Wh = (const float*)d_in[9];
    const float* Uh = (const float*)d_in[10];
    const float* bh = (const float*)d_in[11];
    const float* decW = (const float*)d_in[12];
    const float* decb = (const float*)d_in[13];

    // ---- workspace carve-out (all 256B aligned)
    char* p = (char*)d_ws;
    auto alloc = [&](size_t bytes) { char* r = p; p += (bytes + 255) & ~(size_t)255; return (void*)r; };

    bf16_t* Wxb  = (bf16_t*)alloc(L * HH * sizeof(bf16_t));
    bf16_t* Urb  = (bf16_t*)alloc(L * HH * sizeof(bf16_t));
    bf16_t* Wzb  = (bf16_t*)alloc(L * HH * sizeof(bf16_t));
    bf16_t* Uzb  = (bf16_t*)alloc(L * HH * sizeof(bf16_t));
    bf16_t* Whb  = (bf16_t*)alloc(L * HH * sizeof(bf16_t));
    bf16_t* Uhb  = (bf16_t*)alloc(L * HH * sizeof(bf16_t));
    bf16_t* decWb = (bf16_t*)alloc((size_t)V * H * sizeof(bf16_t));
    bf16_t* Xemb = (bf16_t*)alloc((size_t)SBH * sizeof(bf16_t));
    float*  Ax0  = (float*)alloc((size_t)SBH * sizeof(float));
    float*  Az0  = (float*)alloc((size_t)SBH * sizeof(float));
    float*  Ah0  = (float*)alloc((size_t)SBH * sizeof(float));
    bf16_t* H1all = (bf16_t*)alloc((size_t)SBH * sizeof(bf16_t));
    float*  hf   = (float*)alloc((size_t)L * BH * sizeof(float));
    bf16_t* hb   = (bf16_t*)alloc((size_t)L * BH * sizeof(bf16_t));
    float*  Gr   = (float*)alloc((size_t)BH * sizeof(float));
    float*  Zb   = (float*)alloc((size_t)BH * sizeof(float));
    float*  Hc   = (float*)alloc((size_t)BH * sizeof(float));
    bf16_t* rhb  = (bf16_t*)alloc((size_t)BH * sizeof(bf16_t));

    auto nb = [](int n) { return (n + 255) / 256; };
    auto gemm = [&](const bf16_t* A, const bf16_t* Wm, const float* Ci, const float* bi,
                    float* Co, int M, int N, int K, int act) {
        dim3 grid(N / 64, M / 64);
        gemm_bf16_wmma<<<grid, 128, 0, stream>>>(A, Wm, Ci, bi, Co, M, N, K, act);
    };

    // ---- one-time prep: weight down-conversion, embedding gather, h init
    cvt_f32_bf16<<<nb(L * (int)HH), 256, 0, stream>>>(Wx, Wxb, L * (int)HH);
    cvt_f32_bf16<<<nb(L * (int)HH), 256, 0, stream>>>(Ur, Urb, L * (int)HH);
    cvt_f32_bf16<<<nb(L * (int)HH), 256, 0, stream>>>(Wz, Wzb, L * (int)HH);
    cvt_f32_bf16<<<nb(L * (int)HH), 256, 0, stream>>>(Uz, Uzb, L * (int)HH);
    cvt_f32_bf16<<<nb(L * (int)HH), 256, 0, stream>>>(Wh, Whb, L * (int)HH);
    cvt_f32_bf16<<<nb(L * (int)HH), 256, 0, stream>>>(Uh, Uhb, L * (int)HH);
    cvt_f32_bf16<<<nb(V * H), 256, 0, stream>>>(decW, decWb, V * H);
    embed_gather<<<nb(S * B * E), 256, 0, stream>>>(inputs, emb, Xemb, S * B * E, E);
    init_h<<<nb(L * BH), 256, 0, stream>>>(hidden, hf, hb, L * BH);

    // ---- hoisted layer-0 input projections: [S*B, H] x [H, H]^T  (x3)
    gemm(Xemb, Wxb, nullptr, nullptr, Ax0, S * B, H, H, ACT_NONE);
    gemm(Xemb, Wzb, nullptr, nullptr, Az0, S * B, H, H, ACT_NONE);
    gemm(Xemb, Whb, nullptr, nullptr, Ah0, S * B, H, H, ACT_NONE);

    // ---- sequential scan over time
    float*  hf0 = hf;        float*  hf1 = hf + BH;
    bf16_t* hb0 = hb;        bf16_t* hb1 = hb + BH;
    const float* br0 = br;   const float* br1 = br + H;
    const float* bz0 = bz;   const float* bz1 = bz + H;
    const float* bh0 = bh;   const float* bh1 = bh + H;

    for (int t = 0; t < S; ++t) {
        const size_t off = (size_t)t * BH;

        // ----- layer 0 (x-projections precomputed in Ax0/Az0/Ah0)
        gemm(hb0, Urb, Ax0 + off, br0, Gr, B, H, H, ACT_SIGMOID);   // r
        gemm(hb0, Uzb, Az0 + off, bz0, Zb, B, H, H, ACT_SIGMOID);   // z
        rh_mul<<<nb(BH), 256, 0, stream>>>(Gr, hf0, rhb, BH);
        gemm(rhb, Uhb, Ah0 + off, bh0, Hc, B, H, H, ACT_TANH);      // h~
        gru_update<<<nb(BH), 256, 0, stream>>>(Zb, Hc, hf0, hb0, nullptr, BH);

        // ----- layer 1 (x = fresh h0; x-projections computed here)
        gemm(hb0, Wxb + HH, nullptr, nullptr, Gr, B, H, H, ACT_NONE);
        gemm(hb1, Urb + HH, Gr, br1, Gr, B, H, H, ACT_SIGMOID);     // r
        gemm(hb0, Wzb + HH, nullptr, nullptr, Zb, B, H, H, ACT_NONE);
        gemm(hb1, Uzb + HH, Zb, bz1, Zb, B, H, H, ACT_SIGMOID);     // z
        gemm(hb0, Whb + HH, nullptr, nullptr, Hc, B, H, H, ACT_NONE);
        rh_mul<<<nb(BH), 256, 0, stream>>>(Gr, hf1, rhb, BH);
        gemm(rhb, Uhb + HH, Hc, bh1, Hc, B, H, H, ACT_TANH);        // h~
        gru_update<<<nb(BH), 256, 0, stream>>>(Zb, Hc, hf1, hb1, H1all + off, BH);
    }

    // ---- hoisted decoder: logits[S*B, V] = H1all * decW^T + decb
    gemm(H1all, decWb, nullptr, decb, (float*)d_out, S * B, V, H, ACT_NONE);

    // ---- h_final [L,B,H] appended after logits
    copy_f32<<<nb(L * BH), 256, 0, stream>>>(hf, (float*)d_out + (size_t)S * B * V, L * BH);
}
